// ModulatedDeformConv_62886911148487
// MI455X (gfx1250) — matrix operand order repaired and checked by reading, not compiled, not run on hardware
//
#include <hip/hip_runtime.h>
#include <hip/hip_bf16.h>

typedef _Float16 v8h  __attribute__((ext_vector_type(8)));
typedef _Float16 v16h __attribute__((ext_vector_type(16)));
typedef float    v8f  __attribute__((ext_vector_type(8)));

// 8-byte row-pair sample, honest 4-byte alignment (no misaligned-b64 UB)
struct __attribute__((packed, aligned(4))) fpair { float a, b; };

#define B_     8
#define CIN    128
#define COUT   128
#define H_     64
#define W_     64
#define KK     9
#define KRED   1152          // Cin * 9 reduction length
#define P_     4096          // H*W pixels per image
#define PT     128           // pixel tile per workgroup
#define KC     32            // reduction chunk (WMMA K)
#define NCHUNK (KRED / KC)   // 36
#define LDB    40            // f16 row stride for B tile (16B aligned, padded)

// ---------------------------------------------------------------------------
// One-time prep: weight[Cout,1152] f32 -> f16, pre-swizzled into WMMA A-frag
// lane order [k0..7, k16..23 | k8..15, k24..31]. 294 KB -> L2 resident.
// ---------------------------------------------------------------------------
__global__ void prep_weight_frags(const float* __restrict__ weight,
                                  _Float16* __restrict__ wq)
{
    const int idx = blockIdx.x * 256 + threadIdx.x;     // over 36*128*32
    if (idx >= NCHUNK * COUT * KC) return;
    const int slot = idx & 31;
    const int row  = (idx >> 5) & 127;
    const int ch   = idx >> 12;
    const int grp  = slot >> 3;
    const int kin  = slot & 7;
    const int k    = (grp == 0 ? 0 : grp == 1 ? 16 : grp == 2 ? 8 : 24) + kin;
    wq[idx] = (_Float16)weight[row * KRED + ch * KC + k];
}

// ---------------------------------------------------------------------------
// Sample one 32 x 128 cols chunk into an LDS column. Thread owns pixel pj and
// 16 consecutive kk rows: (c,k) advance incrementally (one div-by-9 / chunk).
// Each element: 1 ds_b128 (weights) + 1 ds_b32 (base) + 2 row-pair gathers.
// ---------------------------------------------------------------------------
__device__ __forceinline__ void sample_chunk(int mb, int kkbase, int pj,
                                             const float* __restrict__ input_b,
                                             const float* __restrict__ s_w,
                                             const int* __restrict__ s_i,
                                             _Float16* __restrict__ dstcol)
{
    int m = mb + kkbase;
    int c = m / 9;                       // once per chunk
    int k = m - c * 9;
    const float* ib = input_b + (size_t)c * P_;
#pragma unroll
    for (int i = 0; i < 16; ++i) {
        const int et = k * PT + pj;                     // tap record index
        const float4 w4 = *(const float4*)(s_w + et * 4);   // ds_load_b128
        const int ibase = s_i[et];                          // ds_load_b32
        const fpair r0 = *(const fpair*)(ib + ibase);       // row yb
        const fpair r1 = *(const fpair*)(ib + ibase + W_);  // row yb+1
        const float v = w4.x * r0.a + w4.y * r0.b
                      + w4.z * r1.a + w4.w * r1.b;
        dstcol[kkbase + i] = (_Float16)v;
        if (++k == KK) { k = 0; ib += P_; }             // next input channel
    }
}

// ---------------------------------------------------------------------------
// Fused deformable-im2col + WMMA GEMM. One WG = 128(Cout) x 128(pixels) tile.
// ---------------------------------------------------------------------------
__launch_bounds__(256, 1)
__global__ void dcnv2_fused_wmma(const float* __restrict__ input,
                                 const float* __restrict__ offset,
                                 const float* __restrict__ mask,
                                 const _Float16* __restrict__ wq,
                                 const float* __restrict__ bias,
                                 float* __restrict__ out)
{
    __shared__ float    s_w[KK * PT * 4];     // pair-weights (mask/valid premul)
    __shared__ int      s_i[KK * PT];         // pair base index (yb*W + xb)
    __shared__ _Float16 s_b[2][PT * LDB];     // double-buffered cols tile [pix][k]

    const int tid = threadIdx.x;
    const int b   = blockIdx.y;
    const int p0  = blockIdx.x * PT;
    const float* __restrict__ input_b = input + (size_t)b * CIN * P_;

    // ---------- stage 0: pair-based bilinear taps (shared across all Cin) ----------
    for (int e = tid; e < KK * PT; e += 256) {
        const int k  = e >> 7;
        const int pj = e & 127;
        const int p  = p0 + pj;
        const int y  = p >> 6;
        const int x  = p & 63;
        const int ky = k / 3;
        const int kx = k - ky * 3;

        const float oy = offset[(((b * 2 * KK) + 2 * k    ) * H_ + y) * W_ + x];
        const float ox = offset[(((b * 2 * KK) + 2 * k + 1) * H_ + y) * W_ + x];
        const float mv = mask  [(((b * KK)     +     k    ) * H_ + y) * W_ + x];

        const float py = oy + (float)(y - 1 + ky);   // stride=1, pad=1, dil=1
        const float px = ox + (float)(x - 1 + kx);
        const float fy = floorf(py);
        const float fx = floorf(px);
        const float ly = py - fy, lx = px - fx;
        const int   y0 = (int)fy, x0 = (int)fx;
        const int   y1 = y0 + 1,  x1 = x0 + 1;

        // clamped pair bases; weights on pair elements via indicators
        // (at most one indicator fires per element; covers borders and
        //  fully-out-of-range exactly like the clamp+validity reference)
        const int yb = min(max(y0, 0), H_ - 2);
        const int xb = min(max(x0, 0), W_ - 2);
        const float wx0 = (x0 == xb     ? 1.f - lx : 0.f) + (x1 == xb     ? lx : 0.f);
        const float wx1 = (x0 == xb + 1 ? 1.f - lx : 0.f) + (x1 == xb + 1 ? lx : 0.f);
        const float wy0 = (y0 == yb     ? 1.f - ly : 0.f) + (y1 == yb     ? ly : 0.f);
        const float wy1 = (y0 == yb + 1 ? 1.f - ly : 0.f) + (y1 == yb + 1 ? ly : 0.f);

        const int ei = e * 4;
        s_w[ei + 0] = mv * wy0 * wx0;
        s_w[ei + 1] = mv * wy0 * wx1;
        s_w[ei + 2] = mv * wy1 * wx0;
        s_w[ei + 3] = mv * wy1 * wx1;
        s_i[e]      = yb * W_ + xb;
    }

    // ---------- wave tiling: 8 waves = 2 (M) x 4 (N); wave tile 64x32 ----------
    const int wave    = tid >> 5;
    const int lane    = tid & 31;
    const int wm      = wave & 1;        // * 64 in M (Cout)
    const int wn      = wave >> 1;       // * 32 in N (pixels)
    const int l16     = lane & 15;
    const int halfsel = lane >> 4;       // lane group 0 / 1

    // sampling-duty mapping: fixed pixel column + 16 consecutive kk rows
    const int s_pj  = tid & 127;
    const int s_kk0 = (tid >> 7) * 16;   // 0 or 16

    v8f acc[4][2] = {};

    __syncthreads();                     // taps ready

    // prologue: sample chunk 0 into buffer 0
    sample_chunk(0, s_kk0, s_pj, input_b, s_w, s_i, &s_b[0][s_pj * LDB]);
    __syncthreads();

    int buf = 0;
    for (int mb = 0; mb < KRED; mb += KC) {
        const int ci = mb >> 5;          // chunk index (KC == 32)

        // A fragments: straight from pre-swizzled f16 weights in L2
        v16h afrag[4];
#pragma unroll
        for (int mt = 0; mt < 4; ++mt) {
            const int row = wm * 64 + mt * 16 + l16;
            const _Float16* wp = wq + ((size_t)ci * COUT + row) * KC + halfsel * 16;
            v8h lo = *(const v8h*)(wp);
            v8h hi = *(const v8h*)(wp + 8);
#pragma unroll
            for (int i = 0; i < 8; ++i) { afrag[mt][i] = lo[i]; afrag[mt][i + 8] = hi[i]; }
            // WGP-scope (default) prefetch of next chunk's A line
            const int cn = (ci + 1 < NCHUNK) ? ci + 1 : ci;
            const _Float16* wpn = wq + ((size_t)cn * COUT + row) * KC;
            asm volatile("global_prefetch_b8 %0, off" :: "v"(wpn) : "memory");
        }

        // B fragments from LDS (pixel-major: two contiguous b128 per frag)
        v16h bfrag[2];
#pragma unroll
        for (int nt = 0; nt < 2; ++nt) {
            const int col = wn * 32 + nt * 16 + l16;
            const int kb  = halfsel * 8;                   // lanes>=16: K=8..15, 24..31
            v8h lo = *(const v8h*)&s_b[buf][col * LDB + kb];
            v8h hi = *(const v8h*)&s_b[buf][col * LDB + kb + 16];
#pragma unroll
            for (int i = 0; i < 8; ++i) { bfrag[nt][i] = lo[i]; bfrag[nt][i + 8] = hi[i]; }
        }

        // issue matrix ops first: XDL pipe overlaps the sampling VALU below
#pragma unroll
        for (int mt = 0; mt < 4; ++mt)
#pragma unroll
            for (int nt = 0; nt < 2; ++nt)
                acc[mt][nt] = __builtin_amdgcn_wmma_f32_16x16x32_f16(
                    /*neg_a=*/false, afrag[mt],
                    /*neg_b=*/false, bfrag[nt],
                    /*c_mod=*/(short)0, acc[mt][nt],
                    /*reuse_a=*/false, /*reuse_b=*/false);

        // sample next chunk into the other buffer (no conflict with compute)
        const int nmb = mb + KC;
        if (nmb < KRED)
            sample_chunk(nmb, s_kk0, s_pj, input_b, s_w, s_i,
                         &s_b[buf ^ 1][s_pj * LDB]);

        __syncthreads();                 // single barrier per K-chunk
        buf ^= 1;
    }

    // ---------- epilogue: C/D layout -> out[b, co, y, x] + bias ----------
#pragma unroll
    for (int mt = 0; mt < 4; ++mt) {
#pragma unroll
        for (int nt = 0; nt < 2; ++nt) {
            const int cobase = wm * 64 + mt * 16 + halfsel * 8;
            const int pp     = p0 + wn * 32 + nt * 16 + l16;
#pragma unroll
            for (int v = 0; v < 8; ++v) {
                const int co = cobase + v;
                out[((size_t)(b * COUT + co)) * P_ + pp] = acc[mt][nt][v] + bias[co];
            }
        }
    }
}

extern "C" void kernel_launch(void* const* d_in, const int* in_sizes, int n_in,
                              void* d_out, int out_size, void* d_ws, size_t ws_size,
                              hipStream_t stream) {
    const float* input  = (const float*)d_in[0];
    const float* offset = (const float*)d_in[1];
    const float* mask   = (const float*)d_in[2];
    const float* weight = (const float*)d_in[3];
    const float* bias   = (const float*)d_in[4];
    float* out = (float*)d_out;
    _Float16* wq = (_Float16*)d_ws;      // 36*128*32 f16 = 294912 B

    const int total = NCHUNK * COUT * KC;
    prep_weight_frags<<<(total + 255) / 256, 256, 0, stream>>>(weight, wq);

    dim3 grid(P_ / PT, B_);              // 32 pixel tiles x 8 batches
    dcnv2_fused_wmma<<<grid, 256, 0, stream>>>(input, offset, mask, wq, bias, out);

    (void)in_sizes; (void)n_in; (void)out_size; (void)ws_size;
}
